// DummyAttention_87505663689046
// MI455X (gfx1250) — matrix operand order repaired and checked by reading, not compiled, not run on hardware
//
#include <hip/hip_runtime.h>
#include <math.h>

#define Bsz 2
#define Ssz 2048
#define Dsz 2048
#define Hn  16
#define HDsz 128

typedef __attribute__((ext_vector_type(16))) __bf16 v16bf;
typedef __attribute__((ext_vector_type(8)))  float  v8f;
typedef __attribute__((ext_vector_type(4)))  unsigned int u32x4;
typedef __attribute__((ext_vector_type(4)))  int i32x4;
typedef __attribute__((ext_vector_type(8)))  int i32x8;
typedef unsigned short u16t;
typedef unsigned int   u32t;

union FragU { uint4 q[2]; v16bf v; };
union Pack8 { uint4 q; u16t h[8]; };

__device__ __forceinline__ u16t f32_to_bf16(float f) {
  u32t u = __builtin_bit_cast(u32t, f);
  u = (u + 0x7FFFu + ((u >> 16) & 1u)) >> 16;   // round-to-nearest-even
  return (u16t)u;
}

// ---- Tensor Data Mover: 2D tile (tileW x tileH, 2-byte elements) -> LDS -----
// D# per CDNA5 ISA: group0 = {flags, lds_addr,
// global_addr, type=2}; group1 = {data_size, tensor dims, tile dims, strides}.
__device__ __forceinline__ void tdm_load_tile_2d(unsigned ldsByteAddr,
                                                 const void* gptr,
                                                 unsigned tileW, unsigned tileH,
                                                 unsigned rowStrideElems) {
  unsigned long long ga = (unsigned long long)(size_t)gptr;
  u32x4 g0;
  g0[0] = 1u;                                    // count=1, user descriptor
  g0[1] = ldsByteAddr;                           // lds_addr[31:0]
  g0[2] = (unsigned)(ga & 0xFFFFFFFFu);          // global_addr[31:0]
  g0[3] = (unsigned)((ga >> 32) & 0x01FFFFFFu)   // global_addr[56:32]
          | (2u << 30);                          // type=2 ("image")
  i32x8 g1;
  g1[0] = 0x00010000;                            // data_size=1 (2B); no multicast
  g1[1] = (int)((tileW & 0xFFFFu) << 16);        // tensor_dim0[15:0]
  g1[2] = (int)((tileW >> 16) | ((tileH & 0xFFFFu) << 16));  // dim0 hi | dim1 lo
  g1[3] = (int)((tileH >> 16) | ((tileW & 0xFFFFu) << 16));  // dim1 hi | tile_dim0
  g1[4] = (int)tileH;                            // tile_dim1 | tile_dim2=0
  g1[5] = (int)rowStrideElems;                   // tensor_dim0_stride[31:0]
  g1[6] = 0;                                     // stride hi | dim1_stride lo
  g1[7] = 0;
  i32x4 z4 = {0, 0, 0, 0};
#if defined(__clang_major__) && (__clang_major__ >= 23)
  i32x8 z8 = {0, 0, 0, 0, 0, 0, 0, 0};
  __builtin_amdgcn_tensor_load_to_lds(g0, g1, z4, z4, z8, 0);
#else
  __builtin_amdgcn_tensor_load_to_lds(g0, g1, z4, z4, 0);
#endif
}

// ---------------------------------------------------------------- cvt f32->bf16
__global__ void cvt_f32_to_bf16(const float* __restrict__ src,
                                u16t* __restrict__ dst, int n4) {
  int stride = gridDim.x * blockDim.x;
  for (int i = blockIdx.x * blockDim.x + threadIdx.x; i < n4; i += stride) {
    float4 f = reinterpret_cast<const float4*>(src)[i];
    unsigned long long pk =
        (unsigned long long)f32_to_bf16(f.x)        |
        ((unsigned long long)f32_to_bf16(f.y) << 16)|
        ((unsigned long long)f32_to_bf16(f.z) << 32)|
        ((unsigned long long)f32_to_bf16(f.w) << 48);
    reinterpret_cast<unsigned long long*>(dst)[i] = pk;
  }
}

// ------------------------------------------------- C = A[M,K] @ W[N,K]^T (bf16 WMMA)
// grid = (M/16, D/512), block = 256 (8 waves). Wave w: 16 rows x 64 cols (4 tiles).
__global__ void __launch_bounds__(256)
linear_bf16_wmma(const u16t* __restrict__ A, const u16t* __restrict__ W,
                 u16t* __restrict__ outB, float* __restrict__ outF, float scale) {
  const int lane = threadIdx.x & 31;
  const int wave = threadIdx.x >> 5;
  const int l15  = lane & 15;
  const int half = lane >> 4;
  const int mBase = blockIdx.x * 16;
  const int nBase = blockIdx.y * 512 + wave * 64;

  v8f acc[4];
  #pragma unroll
  for (int t = 0; t < 4; ++t) acc[t] = (v8f){0,0,0,0,0,0,0,0};

  const size_t aRow = (size_t)(mBase + l15) * Dsz;
  for (int k0 = 0; k0 < Dsz; k0 += 32) {
    // A fragment: lane<16 rows 0..15 hold K {0..7,16..23}; lanes 16..31 hold {8..15,24..31}
    FragU a;
    const u16t* ap = A + aRow + k0 + half * 8;
    a.q[0] = *reinterpret_cast<const uint4*>(ap);
    a.q[1] = *reinterpret_cast<const uint4*>(ap + 16);
    __builtin_prefetch(ap + 32, 0, 1);   // global_prefetch_b8 next K chunk
    #pragma unroll
    for (int t = 0; t < 4; ++t) {
      // B fragment: column n of B == row n of W (half-wave h holds K {8h..8h+7, 16+8h..16+8h+7})
      FragU b;
      const u16t* wp = W + (size_t)(nBase + t * 16 + l15) * Dsz + k0 + half * 8;
      b.q[0] = *reinterpret_cast<const uint4*>(wp);
      b.q[1] = *reinterpret_cast<const uint4*>(wp + 16);
      acc[t] = __builtin_amdgcn_wmma_f32_16x16x32_bf16(
          false, a.v, false, b.v, (short)0, acc[t], false, false);
    }
  }

  #pragma unroll
  for (int t = 0; t < 4; ++t) {
    #pragma unroll
    for (int r = 0; r < 8; ++r) {
      const int row = mBase + r + half * 8;
      const int col = nBase + t * 16 + l15;
      const float vv = acc[t][r] * scale;
      if (outB) outB[(size_t)row * Dsz + col] = f32_to_bf16(vv);
      else      outF[(size_t)row * Dsz + col] = vv;
    }
  }
}

// ------------------------------------------------------------- flash attention
// grid = (S/128, H, B), block = 256. Wave w owns q rows [qBase, qBase+16).
// K tiles arrive via TDM (double-buffered, TENSORcnt); V tiles are staged
// transposed by all threads; P transposes through per-wave LDS.
__global__ void __launch_bounds__(256)
flash_attn_wmma(const u16t* __restrict__ Q, const u16t* __restrict__ K,
                const u16t* __restrict__ V, u16t* __restrict__ ctx) {
  __shared__ u16t Kst[2][32][HDsz];  // TDM-loaded K tiles, row-major, 16 KB
  __shared__ u16t Vt[HDsz][32];      // V tile transposed: [d][key], 8 KB
  __shared__ u16t Pb[8][16 * 32];    // per-wave P tile (16 q x 32 keys), 8 KB

  const int lane = threadIdx.x & 31;
  const int wave = threadIdx.x >> 5;
  const int l15  = lane & 15;
  const int half = lane >> 4;
  const int b = blockIdx.z, h = blockIdx.y;
  const int qBase = blockIdx.x * 128 + wave * 16;
  const size_t headOff = (size_t)h * HDsz;

  const u16t* Kbase = K + (size_t)b * Ssz * Dsz + headOff;  // key rows, stride D

  // Q fragments (scale 1/sqrt(HD) already folded in at projection time)
  const u16t* Qrow = Q + ((size_t)(b * Ssz + qBase + l15)) * Dsz + headOff;
  FragU qf[4];
  #pragma unroll
  for (int i = 0; i < 4; ++i) {
    const u16t* p = Qrow + i * 32 + half * 8;
    qf[i].q[0] = *reinterpret_cast<const uint4*>(p);
    qf[i].q[1] = *reinterpret_cast<const uint4*>(p + 16);
  }

  v8f o[8];
  #pragma unroll
  for (int c = 0; c < 8; ++c) o[c] = (v8f){0,0,0,0,0,0,0,0};
  float mrow[8], lrow[8];
  #pragma unroll
  for (int r = 0; r < 8; ++r) { mrow[r] = -3.0e38f; lrow[r] = 0.0f; }

  // kick off TDM for the first K tile
  if (wave == 0)
    tdm_load_tile_2d((unsigned)(size_t)&Kst[0][0][0], Kbase, HDsz, 32, Dsz);

  for (int kt = 0; kt < Ssz; kt += 32) {
    const int buf = (kt >> 5) & 1;
    if (wave == 0) __builtin_amdgcn_s_wait_tensorcnt(0);
    __syncthreads();   // K[buf] visible to all; prior Vt/Kst[buf^1] readers done

    // cooperative transposed staging of V tile (32 keys x 128 dims) -> Vt[d][key]
    #pragma unroll
    for (int it = 0; it < 2; ++it) {
      const int chunk = threadIdx.x + it * 256;   // 512 chunks of 8 bf16
      const int key = chunk >> 4;
      const int dc  = (chunk & 15) * 8;
      Pack8 pk;
      pk.q = *reinterpret_cast<const uint4*>(
          V + ((size_t)(b * Ssz + kt + key)) * Dsz + headOff + dc);
      #pragma unroll
      for (int j = 0; j < 8; ++j) Vt[dc + j][key] = pk.h[j];
    }

    // async-prefetch next K tile into the other buffer while we compute
    if (wave == 0 && kt + 32 < Ssz)
      tdm_load_tile_2d((unsigned)(size_t)&Kst[buf ^ 1][0][0],
                       Kbase + (size_t)(kt + 32) * Dsz, HDsz, 32, Dsz);
    __syncthreads();   // Vt ready

    // scores: S = Q (16x128) @ K_tile^T (128x32); K B-fragments from LDS
    v8f s0 = (v8f){0,0,0,0,0,0,0,0};
    v8f s1 = (v8f){0,0,0,0,0,0,0,0};
    #pragma unroll
    for (int kk = 0; kk < 2; ++kk) {
      const u16t* Krow = &Kst[buf][kk * 16 + l15][0];
      v8f sa = (v8f){0,0,0,0,0,0,0,0};
      #pragma unroll
      for (int i = 0; i < 4; ++i) {
        FragU bf;
        const u16t* p = Krow + i * 32 + half * 8;
        bf.q[0] = *reinterpret_cast<const uint4*>(p);
        bf.q[1] = *reinterpret_cast<const uint4*>(p + 16);
        sa = __builtin_amdgcn_wmma_f32_16x16x32_bf16(
            false, qf[i].v, false, bf.v, (short)0, sa, false, false);
      }
      if (kk == 0) s0 = sa; else s1 = sa;
    }

    // online softmax: rows live as (VGPR r, half-wave) pairs; reduce across 16 lanes
    float alpha[8];
    #pragma unroll
    for (int r = 0; r < 8; ++r) {
      float mx = fmaxf(s0[r], s1[r]);
      #pragma unroll
      for (int d = 1; d < 16; d <<= 1) mx = fmaxf(mx, __shfl_xor(mx, d, 32));
      const float mn = fmaxf(mrow[r], mx);
      const float a  = __expf(mrow[r] - mn);
      const float e0 = __expf(s0[r] - mn);
      const float e1 = __expf(s1[r] - mn);
      float sm = e0 + e1;
      #pragma unroll
      for (int d = 1; d < 16; d <<= 1) sm += __shfl_xor(sm, d, 32);
      lrow[r] = lrow[r] * a + sm;
      mrow[r] = mn;
      alpha[r] = a;
      s0[r] = e0; s1[r] = e1;
    }
    v8f av;
    #pragma unroll
    for (int r = 0; r < 8; ++r) av[r] = alpha[r];
    #pragma unroll
    for (int c = 0; c < 8; ++c) o[c] *= av;

    // P: D-layout -> A-layout transpose through per-wave LDS
    u16t* pw = &Pb[wave][0];
    #pragma unroll
    for (int r = 0; r < 8; ++r) {
      const int row = r + half * 8;
      pw[row * 32 + l15]      = f32_to_bf16(s0[r]);
      pw[row * 32 + 16 + l15] = f32_to_bf16(s1[r]);
    }
    asm volatile("s_wait_dscnt 0" ::: "memory");
    FragU pf;
    {
      const u16t* p = pw + l15 * 32 + half * 8;
      pf.q[0] = *reinterpret_cast<const uint4*>(p);
      pf.q[1] = *reinterpret_cast<const uint4*>(p + 16);
    }

    // O += P (16x32) @ V_tile (32x128): 8 n-chunks, B fragments from transposed LDS
    #pragma unroll
    for (int c = 0; c < 8; ++c) {
      FragU vb;
      const u16t* p = &Vt[c * 16 + l15][half * 8];
      vb.q[0] = *reinterpret_cast<const uint4*>(p);
      vb.q[1] = *reinterpret_cast<const uint4*>(p + 16);
      o[c] = __builtin_amdgcn_wmma_f32_16x16x32_bf16(
          false, pf.v, false, vb.v, (short)0, o[c], false, false);
    }
  }

  // epilogue: O /= l ; store context in [B,S,H*HD] so wo projection is a plain GEMM
  #pragma unroll
  for (int r = 0; r < 8; ++r) lrow[r] = 1.0f / lrow[r];
  #pragma unroll
  for (int c = 0; c < 8; ++c) {
    #pragma unroll
    for (int r = 0; r < 8; ++r) {
      const int qrow = qBase + r + half * 8;
      const int d = c * 16 + l15;
      ctx[((size_t)(b * Ssz + qrow)) * Dsz + headOff + d] =
          f32_to_bf16(o[c][r] * lrow[r]);
    }
  }
}

// ---------------------------------------------------------------------- driver
extern "C" void kernel_launch(void* const* d_in, const int* in_sizes, int n_in,
                              void* d_out, int out_size, void* d_ws, size_t ws_size,
                              hipStream_t stream) {
  (void)in_sizes; (void)n_in; (void)out_size; (void)ws_size;
  const float* x  = (const float*)d_in[0];
  const float* wq = (const float*)d_in[1];
  const float* wk = (const float*)d_in[2];
  const float* wv = (const float*)d_in[3];
  const float* wo = (const float*)d_in[4];
  float* out = (float*)d_out;

  const size_t nX = (size_t)Bsz * Ssz * Dsz;   // 8,388,608
  const size_t nW = (size_t)Dsz * Dsz;         // 4,194,304
  u16t* xb  = (u16t*)d_ws;
  u16t* wqb = xb  + nX;
  u16t* wkb = wqb + nW;
  u16t* wvb = wkb + nW;
  u16t* wob = wvb + nW;
  u16t* Qb  = wob + nW;
  u16t* Kb  = Qb  + nX;
  u16t* Vb  = Kb  + nX;
  u16t* Cb  = Vb  + nX;   // total 112 MB of workspace

  cvt_f32_to_bf16<<<2048, 256, 0, stream>>>(x,  xb,  (int)(nX / 4));
  cvt_f32_to_bf16<<<1024, 256, 0, stream>>>(wq, wqb, (int)(nW / 4));
  cvt_f32_to_bf16<<<1024, 256, 0, stream>>>(wk, wkb, (int)(nW / 4));
  cvt_f32_to_bf16<<<1024, 256, 0, stream>>>(wv, wvb, (int)(nW / 4));
  cvt_f32_to_bf16<<<1024, 256, 0, stream>>>(wo, wob, (int)(nW / 4));

  const dim3 lg(Bsz * Ssz / 16, Dsz / 512);    // (256, 4)
  const float qscale = 0.08838834764831845f;   // 1/sqrt(HD), folded into Q
  linear_bf16_wmma<<<lg, 256, 0, stream>>>(xb, wqb, Qb, nullptr, qscale);
  linear_bf16_wmma<<<lg, 256, 0, stream>>>(xb, wkb, Kb, nullptr, 1.0f);
  linear_bf16_wmma<<<lg, 256, 0, stream>>>(xb, wvb, Vb, nullptr, 1.0f);

  const dim3 fg(Ssz / 128, Hn, Bsz);           // (16, 16, 2)
  flash_attn_wmma<<<fg, 256, 0, stream>>>(Qb, Kb, Vb, Cb);

  linear_bf16_wmma<<<lg, 256, 0, stream>>>(Cb, wob, nullptr, out, 1.0f);
}
